// BitLinear_65506841199020
// MI455X (gfx1250) — compile-verified, matrix operand order checked by reading, not run on hardware
//
#include <hip/hip_runtime.h>
#include <hip/hip_bf16.h>

// ---------------------------------------------------------------------------
// BitLinear forward on gfx1250 (MI455X, wave32, WMMA, async-to-LDS staging).
// y[M,N] = Xbf16[M,K] * Wbin_bf16[N,K]^T + bias, with
//   Wbin[o,i] = sign(w[o,i]) * max(|scale[(o*K+i)/128]|, 1e-8)   (sign(0)->+1)
// M = B*S = 8192, K = IN = 4096, N = OUT = 11008.
// ---------------------------------------------------------------------------

typedef __attribute__((ext_vector_type(16))) __bf16          v16bf;
typedef __attribute__((ext_vector_type(8)))  float           v8f;
typedef __attribute__((ext_vector_type(8)))  unsigned short  v8us;

#define M_DIM 8192
#define N_DIM 11008
#define K_DIM 4096

#define TILE_M 128
#define TILE_N 128
#define TILE_K 64                       // two K=32 WMMA slabs per LDS buffer
#define LDS_STRIDE 72                   // 64 halfs + 8 pad -> 144B row (16B aligned)
#define KSTEPS (K_DIM / TILE_K)         // 64
#define BUF_BYTES (TILE_M * LDS_STRIDE * 2)   // 18432 B per buffer
#define PIPE_DEPTH 4                    // circular LDS buffers (issue 3 tiles ahead)

static __device__ __forceinline__ unsigned short f32_to_bf16_rne(float f) {
    unsigned int u = __float_as_uint(f);
    u += 0x7FFFu + ((u >> 16) & 1u);   // round to nearest even
    return (unsigned short)(u >> 16);
}

// Low 32 bits of a generic pointer to a __shared__ object == LDS byte address.
static __device__ __forceinline__ unsigned lds_addr32(const void* p) {
    return (unsigned)(unsigned long long)p;
}

// CDNA5 async memory->LDS copy (no VGPR data path, tracked by ASYNCcnt).
static __device__ __forceinline__ void async_load_b128(unsigned lds, unsigned voff,
                                                       const unsigned short* base) {
    asm volatile("global_load_async_to_lds_b128 %0, %1, %2 offset:0"
                 :: "v"(lds), "v"(voff), "s"(base) : "memory");
}
// Async loads retire in order: ASYNCcnt <= 16 means the oldest in-flight tile
// (8 asyncs) has fully landed while two newer tiles may still be in flight.
static __device__ __forceinline__ void wait_asynccnt0()  { asm volatile("s_wait_asynccnt 0x0"  ::: "memory"); }
static __device__ __forceinline__ void wait_asynccnt8()  { asm volatile("s_wait_asynccnt 0x8"  ::: "memory"); }
static __device__ __forceinline__ void wait_asynccnt16() { asm volatile("s_wait_asynccnt 0x10" ::: "memory"); }

// -------- kernel 1: binarize weight into bf16 (sign(w) * group scale) -------
__global__ __launch_bounds__(256)
void quant_w_kernel(const float* __restrict__ w,
                    const float* __restrict__ scale,
                    unsigned short* __restrict__ wb) {
    const int base = (blockIdx.x * 256 + threadIdx.x) * 8;   // 8 elems / thread
    const float s  = fmaxf(fabsf(scale[base >> 7]), 1e-8f);  // all 8 share a group
    const unsigned short sb = f32_to_bf16_rne(s);            // s > 0: sign bit clear
    const float4 a = *(const float4*)(w + base);
    const float4 b = *(const float4*)(w + base + 4);
    v8us r;
    r[0] = (unsigned short)(sb | (a.x < 0.0f ? 0x8000u : 0u));
    r[1] = (unsigned short)(sb | (a.y < 0.0f ? 0x8000u : 0u));
    r[2] = (unsigned short)(sb | (a.z < 0.0f ? 0x8000u : 0u));
    r[3] = (unsigned short)(sb | (a.w < 0.0f ? 0x8000u : 0u));
    r[4] = (unsigned short)(sb | (b.x < 0.0f ? 0x8000u : 0u));
    r[5] = (unsigned short)(sb | (b.y < 0.0f ? 0x8000u : 0u));
    r[6] = (unsigned short)(sb | (b.z < 0.0f ? 0x8000u : 0u));
    r[7] = (unsigned short)(sb | (b.w < 0.0f ? 0x8000u : 0u));
    *(v8us*)(wb + base) = r;
}

// -------- kernel 2: convert x f32 -> bf16 (done once; x then lives in L2) ---
__global__ __launch_bounds__(256)
void cvt_x_kernel(const float* __restrict__ x,
                  unsigned short* __restrict__ xb) {
    const int base = (blockIdx.x * 256 + threadIdx.x) * 8;
    const float4 a = *(const float4*)(x + base);
    const float4 b = *(const float4*)(x + base + 4);
    v8us r;
    r[0] = f32_to_bf16_rne(a.x); r[1] = f32_to_bf16_rne(a.y);
    r[2] = f32_to_bf16_rne(a.z); r[3] = f32_to_bf16_rne(a.w);
    r[4] = f32_to_bf16_rne(b.x); r[5] = f32_to_bf16_rne(b.y);
    r[6] = f32_to_bf16_rne(b.z); r[7] = f32_to_bf16_rne(b.w);
    *(v8us*)(xb + base) = r;
}

// -------- kernel 3: bf16 WMMA GEMM, 128x128 tile, 8 waves ------------------
// Staging: global_load_async_to_lds_b128, 4-deep circular buffering (issue 3
// tiles ahead, steady-state wait ASYNCcnt<=16, tail peeled to 16/8/0);
// 16 WMMAs per barrier; operands via ds_load_b128 pairs per the 16-bit
// WMMA A/B lane layout.
union WmmaOp { v16bf bf; v8us u[2]; };

__global__ __launch_bounds__(256)
void bitlinear_gemm_kernel(const unsigned short* __restrict__ Xb,  // [M,K] bf16
                           const unsigned short* __restrict__ Wb,  // [N,K] bf16
                           const float* __restrict__ bias,         // [N]
                           float* __restrict__ out) {              // [M,N]
    __shared__ __align__(16) unsigned short sA[PIPE_DEPTH][TILE_M * LDS_STRIDE];
    __shared__ __align__(16) unsigned short sB[PIPE_DEPTH][TILE_N * LDS_STRIDE];

    const int tid  = threadIdx.x;
    const int lane = tid & 31;
    const int wave = tid >> 5;
    const int wm   = wave >> 1;      // 0..3 -> 32-row slab
    const int wn   = wave & 1;       // 0..1 -> 64-col slab
    const int m0   = blockIdx.x * TILE_M;
    const int n0   = blockIdx.y * TILE_N;

    // Cooperative staging: 1024 chunks of 16B per tile (128 rows x 128B),
    // 4 per thread per matrix. chunk c: row = c>>3, k offset = (c&7)*8 halfs.
    unsigned voffA[4], voffB[4], ldsA[4], ldsB[4];
#pragma unroll
    for (int j = 0; j < 4; ++j) {
        const int c   = tid + 256 * j;
        const int row = c >> 3;
        const int kc  = (c & 7) * 8;
        voffA[j] = ((unsigned)(m0 + row) * K_DIM + kc) * 2u;   // < 2^31
        voffB[j] = ((unsigned)(n0 + row) * K_DIM + kc) * 2u;   // < 2^31
        ldsA[j]  = lds_addr32(&sA[0][row * LDS_STRIDE + kc]);
        ldsB[j]  = lds_addr32(&sB[0][row * LDS_STRIDE + kc]);
    }

    auto stage = [&](int buf, int ksi) {
        const unsigned ko = (unsigned)ksi * (TILE_K * 2u);
        const unsigned bo = (unsigned)buf * BUF_BYTES;
#pragma unroll
        for (int j = 0; j < 4; ++j) {
            async_load_b128(ldsA[j] + bo, voffA[j] + ko, Xb);
            async_load_b128(ldsB[j] + bo, voffB[j] + ko, Wb);
        }
    };

    // WMMA 16-bit operand per-lane addressing (A 16x32 / B 32x16 layouts):
    // lane row/col = lane&15, K base = (lane>>4)*8; VGPRs 0-3 hold K..K+7,
    // VGPRs 4-7 hold K+16..K+23  -> two 16-byte LDS loads per operand.
    const int lrow = lane & 15;
    const int lk   = (lane >> 4) * 8;

    v8f acc[2][4] = {};

    // 16 WMMAs on buffer `cur` (two K=32 slabs).
    auto compute_step = [&](int cur) __attribute__((always_inline)) {
#pragma unroll
        for (int h = 0; h < 2; ++h) {
            const int lkh = h * 32 + lk;
            WmmaOp opA[2], opB[4];
#pragma unroll
            for (int t = 0; t < 2; ++t) {
                const unsigned short* p =
                    &sA[cur][(wm * 32 + t * 16 + lrow) * LDS_STRIDE + lkh];
                opA[t].u[0] = *(const v8us*)p;
                opA[t].u[1] = *(const v8us*)(p + 16);
            }
#pragma unroll
            for (int u = 0; u < 4; ++u) {
                const unsigned short* p =
                    &sB[cur][(wn * 64 + u * 16 + lrow) * LDS_STRIDE + lkh];
                opB[u].u[0] = *(const v8us*)p;
                opB[u].u[1] = *(const v8us*)(p + 16);
            }
#pragma unroll
            for (int t = 0; t < 2; ++t)
#pragma unroll
                for (int u = 0; u < 4; ++u)
                    acc[t][u] = __builtin_amdgcn_wmma_f32_16x16x32_bf16(
                        false, opA[t].bf, false, opB[u].bf,
                        (short)0, acc[t][u], false, false);
        }
    };

    // prologue: stage tiles 0..2 into buffers 0..2 (24 asyncs in flight)
    stage(0, 0);
    stage(1, 1);
    stage(2, 2);

    // steady state: branch-free body (wait16 / barrier / stage / compute)
#pragma unroll 1
    for (int ks = 0; ks < KSTEPS - 3; ++ks) {
        wait_asynccnt16();            // tile ks landed; ks+1, ks+2 may fly
        __syncthreads();              // all waves: tile-ks written, ks-1 reads done
        stage((ks + 3) & (PIPE_DEPTH - 1), ks + 3);
        compute_step(ks & (PIPE_DEPTH - 1));
    }
    // drain: last three tiles with exact thresholds and constant buffers
    wait_asynccnt16(); __syncthreads(); compute_step((KSTEPS - 3) & (PIPE_DEPTH - 1));
    wait_asynccnt8();  __syncthreads(); compute_step((KSTEPS - 2) & (PIPE_DEPTH - 1));
    wait_asynccnt0();  __syncthreads(); compute_step((KSTEPS - 1) & (PIPE_DEPTH - 1));

    // epilogue: C/D layout -> VGPR r holds row (r + (lane>>4)*8), col = lane&15
#pragma unroll
    for (int t = 0; t < 2; ++t) {
        const int rbase = m0 + wm * 32 + t * 16 + (lane >> 4) * 8;
#pragma unroll
        for (int u = 0; u < 4; ++u) {
            const int col = n0 + wn * 64 + u * 16 + (lane & 15);
            const float bv = bias[col];
#pragma unroll
            for (int r = 0; r < 8; ++r)
                out[(size_t)(rbase + r) * N_DIM + col] = acc[t][u][r] + bv;
        }
    }
}

extern "C" void kernel_launch(void* const* d_in, const int* in_sizes, int n_in,
                              void* d_out, int out_size, void* d_ws, size_t ws_size,
                              hipStream_t stream) {
    const float* x     = (const float*)d_in[0];   // [4,2048,4096]
    const float* w     = (const float*)d_in[1];   // [11008,4096]
    const float* bias  = (const float*)d_in[2];   // [11008]
    const float* scale = (const float*)d_in[3];   // [352256]

    unsigned short* Xb = (unsigned short*)d_ws;                   // 64 MB
    unsigned short* Wb = Xb + (size_t)M_DIM * K_DIM;              // 86 MB

    // x: 33,554,432 elems / 8 per thread / 256 = 16384 blocks
    cvt_x_kernel<<<16384, 256, 0, stream>>>(x, Xb);
    // w: 45,088,768 elems / 8 per thread / 256 = 22016 blocks
    quant_w_kernel<<<22016, 256, 0, stream>>>(w, scale, Wb);

    dim3 grid(M_DIM / TILE_M, N_DIM / TILE_N);   // 64 x 86
    bitlinear_gemm_kernel<<<grid, 256, 0, stream>>>(Xb, Wb, bias, (float*)d_out);
}
